// Attn2d_3582002725318
// MI455X (gfx1250) — matrix operand order repaired and checked by reading, not compiled, not run on hardware
//
#include <hip/hip_runtime.h>
#include <stdint.h>

typedef __attribute__((ext_vector_type(16))) _Float16 v16h;
typedef __attribute__((ext_vector_type(8)))  _Float16 v8h;
typedef __attribute__((ext_vector_type(8)))  float    v8f;
typedef __attribute__((ext_vector_type(4)))  unsigned u32x4;
typedef __attribute__((ext_vector_type(8)))  int      i32x8;
typedef __attribute__((ext_vector_type(4)))  int      i32x4;

#define CIN   256
#define HW    4096
#define NOUT  768
#define TM    64
#define TN    64
#define KC    128
#define LDH   136   // padded LDS row stride in halves (272B = 17*16B): 64 data dwords + 4 pad dwords

#if defined(__has_builtin)
# if __has_builtin(__builtin_amdgcn_tensor_load_to_lds) && __has_builtin(__builtin_amdgcn_s_wait_tensorcnt)
#  define HAVE_TDM 1
# endif
#endif
#ifndef HAVE_TDM
# define HAVE_TDM 0
#endif

#if __has_include(<hip/amd_detail/amd_gfx1250_TDM.h>)
# define TDM_6ARG 1
#else
# define TDM_6ARG 0
#endif

__device__ __forceinline__ unsigned pack_h2(float a, float b) {
    _Float16 h0 = (_Float16)a, h1 = (_Float16)b;
    unsigned short u0, u1;
    __builtin_memcpy(&u0, &h0, 2);
    __builtin_memcpy(&u1, &h1, 2);
    return (unsigned)u0 | ((unsigned)u1 << 16);
}

#if HAVE_TDM
// 2-D TDM load: tile_rows rows x tile_k f16 elements (K-contiguous), row stride
// row_stride_elts in memory; LDS gets 4 pad dwords after every 64 data dwords
// (one 128-half row) -> padded row stride of LDH=136 halves, matching the
// WMMA fragment loaders. D# bitfields per CDNA5 ISA section 8.3/8.4.
__device__ __forceinline__ void tdm_load_2d(unsigned lds_off, const void* gptr,
                                            unsigned tile_k, unsigned tile_rows,
                                            unsigned row_stride_elts)
{
    unsigned long long ga = (unsigned long long)(uintptr_t)gptr;
    u32x4 g0;
    g0[0] = 1u;                                   // count=1, user descriptor
    g0[1] = lds_off;                              // lds_addr (bytes)
    g0[2] = (unsigned)(ga & 0xFFFFFFFFu);         // global_addr[31:0]
    g0[3] = (unsigned)((ga >> 32) & 0x01FFFFFFu)  // global_addr[56:32]
          | (2u << 30);                           // type = 2 ("image")
    i32x8 g1;
    g1[0] = (int)((1u << 16)                      // data_size = 2 bytes
                | (1u << 20)                      // pad_enable
                | (5u << 22)                      // pad_interval: 64 dwords
                | (3u << 25));                    // pad_amount: 4 dwords
    g1[1] = (int)((tile_k & 0xFFFFu) << 16);                    // tensor_dim0[15:0]
    g1[2] = (int)(((tile_k >> 16) & 0xFFFFu)
                | ((tile_rows & 0xFFFFu) << 16));               // tensor_dim0 hi | tensor_dim1 lo
    g1[3] = (int)(((tile_rows >> 16) & 0xFFFFu)
                | ((tile_k & 0xFFFFu) << 16));                  // tensor_dim1 hi | tile_dim0
    g1[4] = (int)(tile_rows & 0xFFFFu);                         // tile_dim1 (tile_dim2 = 0 -> 2D)
    g1[5] = (int)row_stride_elts;                               // tensor_dim0_stride lo
    g1[6] = 0;                                                  // stride0 hi | stride1 lo
    g1[7] = 0;                                                  // stride1 hi
    i32x4 g2; g2[0] = 1; g2[1] = 0; g2[2] = 0; g2[3] = 0;       // tensor_dim2 = 1 (benign)
    i32x4 g3; g3[0] = 0; g3[1] = 0; g3[2] = 0; g3[3] = 0;
#if TDM_6ARG
    i32x8 g4; g4[0]=0; g4[1]=0; g4[2]=0; g4[3]=0; g4[4]=0; g4[5]=0; g4[6]=0; g4[7]=0;
    __builtin_amdgcn_tensor_load_to_lds(g0, g1, g2, g3, g4, 0);
#else
    __builtin_amdgcn_tensor_load_to_lds(g0, g1, g2, g3, 0);
#endif
}
#endif

// ---- pre-pass: w (768x256 f32) -> wh f16, same layout ----
__global__ __launch_bounds__(256)
void convert_w_kernel(const float* __restrict__ w, _Float16* __restrict__ wh)
{
    int i = blockIdx.x * 256 + threadIdx.x;     // pair index, 98304 pairs
    float2 v = ((const float2*)w)[i];
    ((unsigned*)wh)[i] = pack_h2(v.x, v.y);
}

// ---- pre-pass: x (b,c,hw f32) -> xT (b,hw,c f16): K-contiguous rows for TDM ----
#define TLD 260
__global__ __launch_bounds__(256)
void transpose_x_kernel(const float* __restrict__ x, _Float16* __restrict__ xT)
{
    __shared__ _Float16 T[64][TLD];
    const int b = blockIdx.y, hw0 = blockIdx.x * 64, tid = threadIdx.x;
    const float* xb = x + (size_t)b * CIN * HW;
    #pragma unroll
    for (int it = 0; it < 32; ++it) {
        int n = tid & 63;
        int c = it * 8 + (tid >> 6) * 2;
        float f0 = xb[(size_t)c       * HW + hw0 + n];
        float f1 = xb[(size_t)(c + 1) * HW + hw0 + n];
        *(unsigned*)&T[n][c] = pack_h2(f0, f1);
    }
    __syncthreads();
    unsigned* dst = (unsigned*)(xT + ((size_t)b * HW + hw0) * CIN);
    #pragma unroll
    for (int it = 0; it < 32; ++it) {
        int r  = it * 2 + (tid >> 7);
        int cp = tid & 127;
        dst[(size_t)r * 128 + cp] = *(const unsigned*)&T[r][cp * 2];
    }
}

// qkv[b] (768 x 4096) = W (768x256) * x[b] (256x4096) + bias, scale folded into
// rows [0,512). Tiles staged by the Tensor Data Mover; compute is ds_load+WMMA.
__global__ __launch_bounds__(256)
void qkv_gemm_kernel(const _Float16* __restrict__ wh, const _Float16* __restrict__ xT,
                     const float* __restrict__ bias, float* __restrict__ qkv)
{
    __shared__ __align__(16) _Float16 smem[2][TM][LDH];  // [0]=A (m,k)  [1]=B (n,k)

    const int tid  = threadIdx.x;
    const int lane = tid & 31;
    const int wave = tid >> 5;
    const int b    = blockIdx.z;
    const int mA   = blockIdx.y * TM;
    const int hw0  = blockIdx.x * TN;

    const int wm = wave >> 1;
    const int wn = wave & 1;
    const int m0 = wm * 16;
    const int n0 = wn * 32;
    const int l15 = lane & 15;
    const int lhi = lane >> 4;

    v8f acc0 = {};
    v8f acc1 = {};

#if HAVE_TDM
    const unsigned lds_base = (unsigned)(uintptr_t)(void*)&smem[0][0][0];
#endif

    for (int ks = 0; ks < CIN; ks += KC) {
#if HAVE_TDM
        if (wave == 0) {
            tdm_load_2d(lds_base,
                        wh + (size_t)mA * CIN + ks, KC, TM, CIN);
            tdm_load_2d(lds_base + (unsigned)(TM * LDH * 2),
                        xT + ((size_t)b * HW + hw0) * CIN + ks, KC, TN, CIN);
            __builtin_amdgcn_s_wait_tensorcnt(0);
        }
#else
        #pragma unroll
        for (int it = 0; it < 16; ++it) {
            int u = it * 256 + tid;                    // 4096 dwords of A
            int m = u >> 6, d = u & 63;
            *(unsigned*)&smem[0][m][d * 2] =
                ((const unsigned*)wh)[(((size_t)(mA + m) * CIN) + ks) / 2 + d];
        }
        #pragma unroll
        for (int it = 0; it < 16; ++it) {
            int u = it * 256 + tid;                    // 4096 dwords of B
            int n = u >> 6, d = u & 63;
            *(unsigned*)&smem[1][n][d * 2] =
                ((const unsigned*)xT)[(((size_t)b * HW + hw0 + n) * CIN + ks) / 2 + d];
        }
#endif
        __syncthreads();

        #pragma unroll
        for (int kk = 0; kk < KC; kk += 32) {
            // A fragment: 16 consecutive K per lane; low lanes kk..+15, high +16
            int arow = m0 + l15;
            int akb  = kk + (lhi << 4);
            v8h a0 = *(const v8h*)(&smem[0][arow][akb]);
            v8h a1 = *(const v8h*)(&smem[0][arow][akb + 8]);
            v16h af = __builtin_shufflevector(a0, a1, 0,1,2,3,4,5,6,7,8,9,10,11,12,13,14,15);

            // B fragments: 16 consecutive K per lane; low lanes kk..+15, high +16
            int bkb = kk + (lhi << 4);
            v8h b00 = *(const v8h*)(&smem[1][n0 + l15][bkb]);
            v8h b01 = *(const v8h*)(&smem[1][n0 + l15][bkb + 8]);
            v16h bf0 = __builtin_shufflevector(b00, b01, 0,1,2,3,4,5,6,7,8,9,10,11,12,13,14,15);
            v8h b10 = *(const v8h*)(&smem[1][n0 + 16 + l15][bkb]);
            v8h b11 = *(const v8h*)(&smem[1][n0 + 16 + l15][bkb + 8]);
            v16h bf1 = __builtin_shufflevector(b10, b11, 0,1,2,3,4,5,6,7,8,9,10,11,12,13,14,15);

            acc0 = __builtin_amdgcn_wmma_f32_16x16x32_f16(false, af, false, bf0,
                                                          (short)0, acc0, false, false);
            acc1 = __builtin_amdgcn_wmma_f32_16x16x32_f16(false, af, false, bf1,
                                                          (short)0, acc1, false, false);
        }
        __syncthreads();
    }

    // epilogue: C layout VGPR r -> M = r + 8*lhi, N = lane&15
    const float kscale = 0.42044820762685725f;   // 32^-0.25
    float* outb = qkv + (size_t)b * NOUT * HW;
    #pragma unroll
    for (int r = 0; r < 8; ++r) {
        int gm   = mA + m0 + r + (lhi << 3);
        float bv = bias[gm];
        float sc = (gm < 512) ? kscale : 1.0f;
        outb[(size_t)gm * HW + hw0 + n0 + l15]      = (acc0[r] + bv) * sc;
        outb[(size_t)gm * HW + hw0 + n0 + 16 + l15] = (acc1[r] + bv) * sc;
    }
}

// 3x3 local attention: thread = (pixel, head). q,k rows in qkv are pre-scaled.
__global__ __launch_bounds__(256)
void attn_kernel(const float* __restrict__ qkv, const float* __restrict__ pos,
                 float* __restrict__ out)
{
    __shared__ float posL[256 * 9];
    const int tid = threadIdx.y * 32 + threadIdx.x;
    for (int i = tid; i < 256 * 9; i += 256) posL[i] = pos[i];
    __syncthreads();

    const int b  = blockIdx.y;
    const int hw = blockIdx.x * 32 + threadIdx.x;
    const int h  = hw >> 6;
    const int w0 = hw & 63;
    const int cb = threadIdx.y * 32;
    const float* base = qkv + (size_t)b * NOUT * HW;

    float q[32];
    #pragma unroll
    for (int c = 0; c < 32; ++c) q[c] = base[(size_t)(cb + c) * HW + hw];

    float lg[9];
    #pragma unroll
    for (int k = 0; k < 9; ++k) {
        float s = 0.f;
        #pragma unroll
        for (int c = 0; c < 32; ++c) s += q[c] * posL[(cb + c) * 9 + k];
        lg[k] = s;
    }
    #pragma unroll
    for (int k = 0; k < 9; ++k) {
        int di = k / 3 - 1, dj = k % 3 - 1;
        int hh = h + di, ww = w0 + dj;
        if (hh >= 0 && hh < 64 && ww >= 0 && ww < 64) {
            const float* kp = base + (size_t)(256 + cb) * HW + hw + di * 64 + dj;
            float s = 0.f;
            #pragma unroll
            for (int c = 0; c < 32; ++c) s += q[c] * kp[(size_t)c * HW];
            lg[k] += s;
        }
    }

    float mx = lg[0];
    #pragma unroll
    for (int k = 1; k < 9; ++k) mx = fmaxf(mx, lg[k]);
    float se = 0.f;
    #pragma unroll
    for (int k = 0; k < 9; ++k) { lg[k] = __expf(lg[k] - mx); se += lg[k]; }
    const float inv = 1.f / se;

    float acc[32];
    #pragma unroll
    for (int c = 0; c < 32; ++c) acc[c] = 0.f;
    #pragma unroll
    for (int k = 0; k < 9; ++k) {
        int di = k / 3 - 1, dj = k % 3 - 1;
        int hh = h + di, ww = w0 + dj;
        if (hh >= 0 && hh < 64 && ww >= 0 && ww < 64) {
            float a = lg[k] * inv;
            const float* vp = base + (size_t)(512 + cb) * HW + hw + di * 64 + dj;
            #pragma unroll
            for (int c = 0; c < 32; ++c) acc[c] += a * vp[(size_t)c * HW];
        }
    }
    #pragma unroll
    for (int c = 0; c < 32; ++c)
        out[((size_t)b * 256 + cb + c) * HW + hw] = acc[c];
}

extern "C" void kernel_launch(void* const* d_in, const int* in_sizes, int n_in,
                              void* d_out, int out_size, void* d_ws, size_t ws_size,
                              hipStream_t stream) {
    (void)in_sizes; (void)n_in; (void)out_size; (void)ws_size;
    const float* x    = (const float*)d_in[0];   // (4,256,64,64)
    const float* w    = (const float*)d_in[1];   // (768,256)
    const float* bias = (const float*)d_in[2];   // (768,)
    const float* pos  = (const float*)d_in[3];   // (256,9)
    float* out = (float*)d_out;                  // (4,256,64,64)

    // workspace: qkv f32 (50.3MB) | xT f16 (8.4MB) | wh f16 (0.4MB)
    float*    qkv = (float*)d_ws;
    _Float16* xT  = (_Float16*)((char*)d_ws + 50331648);
    _Float16* wh  = (_Float16*)((char*)d_ws + 58720256);

    convert_w_kernel <<<dim3(384),        256,        0, stream>>>(w, wh);
    transpose_x_kernel<<<dim3(64, 4),     256,        0, stream>>>(x, xT);
    qkv_gemm_kernel  <<<dim3(64, 12, 4),  256,        0, stream>>>(wh, xT, bias, qkv);
    attn_kernel      <<<dim3(128, 4),     dim3(32, 8), 0, stream>>>(qkv, pos, out);
}
